// MultiHeadAttention_46067819217243
// MI455X (gfx1250) — matrix operand order repaired and checked by reading, not compiled, not run on hardware
//
#include <hip/hip_runtime.h>

typedef __attribute__((ext_vector_type(16))) __bf16 v16bf;
typedef __attribute__((ext_vector_type(8)))  float  v8f;
typedef unsigned int v4u __attribute__((ext_vector_type(4)));
typedef int v8i __attribute__((ext_vector_type(8)));
typedef int v4i __attribute__((ext_vector_type(4)));

#define S_LEN 2048
#define HDIM  1024      // H * Dk == D
#define NHEADS 16
#define DK 64

union Frag {
    v16bf v;
    unsigned int u[8];
};

__device__ __forceinline__ v8f wmma_bf16(const Frag& a, const Frag& b, v8f c) {
    // D = A(16x32 bf16) x B(32x16 bf16) + C(16x16 f32)
    return __builtin_amdgcn_wmma_f32_16x16x32_bf16(
        false, a.v, false, b.v, (short)0, c, false, false);
}

// ---------------------------------------------------------------------------
// TDM: async 2D bf16 tile load, global -> LDS.  Row length 64 elements
// (128 B = 32 dwords), LDS padded by 4 dwords per row => stride 72 bf16.
// D# bitfields per CDNA5 ISA (group0 128b, group1 256b).
// This toolchain's builtin takes 6 args (g0, g1, g2, g3, g4, cpol).
// ---------------------------------------------------------------------------
__device__ __forceinline__ void tdm_load_tile_bf16(
    const void* gaddr, unsigned lds_off,
    unsigned tile_rows, unsigned tensor_rows, unsigned row_stride_elems)
{
    const unsigned row_elems = 64;               // tile_dim0 == tensor_dim0
    unsigned long long ga = (unsigned long long)gaddr;
    v4u g0;
    g0.x = 1u;                                   // count=1, user descriptor
    g0.y = lds_off;                              // lds_addr (bytes)
    g0.z = (unsigned)(ga & 0xffffffffu);         // global_addr[31:0]
    g0.w = (unsigned)((ga >> 32) & 0x01ffffffu)  // global_addr[56:32]
         | (2u << 30);                           // type = 2 ("image")
    v8i g1;
    // data_size=1 (2B) | pad_enable | pad_interval=4 (32 dw) | pad_amount=3 (4 dw)
    g1[0] = (int)((1u << 16) | (1u << 20) | (4u << 22) | (3u << 25));
    g1[1] = (int)(row_elems << 16);              // tensor_dim0[15:0] @ bits[63:48]
    g1[2] = (int)(tensor_rows << 16);            // dim0 hi=0, tensor_dim1 lo
    g1[3] = (int)(row_elems << 16);              // dim1 hi=0, tile_dim0
    g1[4] = (int)(tile_rows & 0xffffu);          // tile_dim1 (tile_dim2=0)
    g1[5] = (int)row_stride_elems;               // tensor_dim0_stride[31:0]
    g1[6] = 0;                                   // stride hi, dim1_stride lo
    g1[7] = 0;
    v4i z4 = {0, 0, 0, 0};                       // 2D tensor: groups 2/3 unused
    v8i z8 = {0, 0, 0, 0, 0, 0, 0, 0};
    __builtin_amdgcn_tensor_load_to_lds(g0, g1, z4, z4, z8, 0);
}

// ---------------------------------------------------------------------------
// GEMM: Out(MxN) = A(MxK) * W(KxN) + bias.  W, bias always fp32.
// A is fp32 (A_BF16=false) or bf16 (true); Out is bf16 or fp32.
// Block tile 128x128, K-tile 64. 8 waves: 4 (M) x 2 (N), wave tile 32x64.
// (fp32->bf16 conversion happens while staging, so TDM is not usable here.)
// ---------------------------------------------------------------------------
template<bool A_BF16, bool OUT_BF16>
__global__ __launch_bounds__(256)
void gemm_bias_kernel(const void* __restrict__ Ap, const float* __restrict__ W,
                      const float* __restrict__ bias, void* __restrict__ Out,
                      int M, int N, int K)
{
    __shared__ __bf16 As[128][72];   // row-major A tile (pad 72 vs 64)
    __shared__ __bf16 Bt[128][72];   // W tile, stored transposed: Bt[n][k]

    const int tid  = threadIdx.x;
    const int lane = tid & 31;
    const int wave = tid >> 5;
    const int hh   = lane >> 4;    // 16-lane half
    const int l16  = lane & 15;

    const int m0 = blockIdx.y * 128;
    const int n0 = blockIdx.x * 128;
    const int rowBase = (wave >> 1) * 32;   // wave M offset in tile
    const int colBase = (wave & 1) * 64;    // wave N offset in tile

    v8f acc[2][4];
    const v8f vzero = {0.f,0.f,0.f,0.f,0.f,0.f,0.f,0.f};
#pragma unroll
    for (int i = 0; i < 2; ++i)
#pragma unroll
        for (int j = 0; j < 4; ++j) acc[i][j] = vzero;

    for (int k0 = 0; k0 < K; k0 += 64) {
        // ---- stage A tile 128x64 (convert to bf16 if needed) ----
        if (A_BF16) {
            const __bf16* A = (const __bf16*)Ap;
#pragma unroll
            for (int i = 0; i < 8; ++i) {
                int idx = tid + i * 256;
                int r = idx >> 4, c4 = (idx & 15) * 4;
                const unsigned int* src =
                    (const unsigned int*)(A + (size_t)(m0 + r) * K + k0 + c4);
                *(unsigned int*)&As[r][c4]     = src[0];
                *(unsigned int*)&As[r][c4 + 2] = src[1];
            }
        } else {
            const float* A = (const float*)Ap;
#pragma unroll
            for (int i = 0; i < 8; ++i) {
                int idx = tid + i * 256;
                int r = idx >> 4, c4 = (idx & 15) * 4;
                const float4 f = *(const float4*)(A + (size_t)(m0 + r) * K + k0 + c4);
                As[r][c4 + 0] = (__bf16)f.x;  As[r][c4 + 1] = (__bf16)f.y;
                As[r][c4 + 2] = (__bf16)f.z;  As[r][c4 + 3] = (__bf16)f.w;
            }
        }
        // ---- stage W tile 64x128, transposed into Bt[n][k] ----
#pragma unroll
        for (int i = 0; i < 8; ++i) {
            int idx = tid + i * 256;
            int r = idx >> 5, c4 = (idx & 31) * 4;  // r: k-row 0..63
            const float4 f = *(const float4*)(W + (size_t)(k0 + r) * N + n0 + c4);
            Bt[c4 + 0][r] = (__bf16)f.x;  Bt[c4 + 1][r] = (__bf16)f.y;
            Bt[c4 + 2][r] = (__bf16)f.z;  Bt[c4 + 3][r] = (__bf16)f.w;
        }
        if (k0 + 64 < K) {  // speculative prefetch of next W tile
            __builtin_prefetch((const char*)(W + (size_t)(k0 + 64) * N + n0), 0, 1);
        }
        __syncthreads();

#pragma unroll
        for (int kt = 0; kt < 2; ++kt) {           // two K=32 WMMA steps
            Frag fa[2], fb[4];
#pragma unroll
            for (int mt = 0; mt < 2; ++mt) {       // A frags (16x32, M=l16)
                int r = rowBase + mt * 16 + l16;
#pragma unroll
                for (int vv = 0; vv < 8; ++vv) {
                    int kk = kt * 32 + (hh << 4) + (vv << 1);
                    fa[mt].u[vv] = *(const unsigned int*)&As[r][kk];
                }
            }
#pragma unroll
            for (int nt = 0; nt < 4; ++nt) {       // B frags (32x16, N=l16)
                int n = colBase + nt * 16 + l16;
#pragma unroll
                for (int vv = 0; vv < 8; ++vv) {
                    int kk = kt * 32 + (hh << 4) + (vv << 1);
                    fb[nt].u[vv] = *(const unsigned int*)&Bt[n][kk];
                }
            }
#pragma unroll
            for (int mt = 0; mt < 2; ++mt)
#pragma unroll
                for (int nt = 0; nt < 4; ++nt)
                    acc[mt][nt] = wmma_bf16(fa[mt], fb[nt], acc[mt][nt]);
        }
        __syncthreads();
    }

    // ---- epilogue: bias + store (C layout: row = r + 8*hh, col = l16) ----
#pragma unroll
    for (int mt = 0; mt < 2; ++mt) {
#pragma unroll
        for (int nt = 0; nt < 4; ++nt) {
            int col = n0 + colBase + nt * 16 + l16;
            float bv = bias[col];
#pragma unroll
            for (int r = 0; r < 8; ++r) {
                int row = m0 + rowBase + mt * 16 + r + 8 * hh;
                float val = acc[mt][nt][r] + bv;
                if (OUT_BF16)
                    ((__bf16*)Out)[(size_t)row * N + col] = (__bf16)val;
                else
                    ((float*)Out)[(size_t)row * N + col] = val;
            }
        }
    }
}

// ---------------------------------------------------------------------------
// Flash attention: one WG per (b, head, 128-query block). 8 waves, each owns
// 16 query rows. Key block = 64 (halves softmax/barrier cost per WMMA vs 32).
// Q and K tiles staged by the Tensor Data Mover (async, TENSORcnt); V staged
// manually (needs transpose). qh/kh/vh: (B,S,H,Dk) bf16. mask: (B,1,S) i32.
// ---------------------------------------------------------------------------
__global__ __launch_bounds__(256)
void flash_attn_kernel(const __bf16* __restrict__ qh, const __bf16* __restrict__ kh,
                       const __bf16* __restrict__ vh, const int* __restrict__ mask,
                       __bf16* __restrict__ concat)
{
    __shared__ __bf16 Qs[128][72];       // Q tile (TDM, pad => stride 72)
    __shared__ __bf16 Ks[64][72];        // K block (TDM, pad => stride 72)
    __shared__ __bf16 Vt[64][72];        // V block transposed: Vt[dk][key]
    __shared__ __bf16 Ps[8][16][72];     // per-wave P bounce (C->A relayout)

    const int tid  = threadIdx.x;
    const int lane = tid & 31;
    const int wave = tid >> 5;
    const int hh   = lane >> 4;
    const int l16  = lane & 15;

    const int qb = blockIdx.x;           // query block (16)
    const int hd = blockIdx.y;           // head (16)
    const int b  = blockIdx.z;           // batch (4)

    const unsigned qs_base = (unsigned)(size_t)&Qs[0][0];
    const unsigned ks_base = (unsigned)(size_t)&Ks[0][0];

    // ---- stage Q tile 128x64 via TDM ----
    if (wave == 0) {
        tdm_load_tile_bf16(qh + (size_t)(b * S_LEN + qb * 128) * HDIM + hd * DK,
                           qs_base, 128, 128, HDIM);
        __builtin_amdgcn_s_wait_tensorcnt(0);
    }
    __syncthreads();

    // Q fragments resident in registers: rows wave*16 .. +15, Dk = 2 x K32
    Frag qa[2];
    {
        int r = wave * 16 + l16;
#pragma unroll
        for (int kt = 0; kt < 2; ++kt)
#pragma unroll
            for (int vv = 0; vv < 8; ++vv) {
                int kk = kt * 32 + (hh << 4) + (vv << 1);
                qa[kt].u[vv] = *(const unsigned int*)&Qs[r][kk];
            }
    }

    float m_i[8], l_i[8];
    v8f o[4];
    const v8f vzero = {0.f,0.f,0.f,0.f,0.f,0.f,0.f,0.f};
#pragma unroll
    for (int r = 0; r < 8; ++r) { m_i[r] = -1e30f; l_i[r] = 0.f; }
#pragma unroll
    for (int t = 0; t < 4; ++t) o[t] = vzero;

    for (int kb = 0; kb < S_LEN / 64; ++kb) {
        __syncthreads();                 // previous block's compute done
        // ---- K block 64x64 via TDM (async while we stage V manually) ----
        if (wave == 0) {
            tdm_load_tile_bf16(kh + (size_t)(b * S_LEN + kb * 64) * HDIM + hd * DK,
                               ks_base, 64, 64, HDIM);
        }
        // ---- V block transposed: Vt[dk][key], 64x64 ----
#pragma unroll
        for (int i = 0; i < 8; ++i) {
            int idx = tid + i * 256;     // 0..2047 dwords
            int r = idx >> 5, c2 = idx & 31;
            const unsigned int* src = (const unsigned int*)
                (vh + (size_t)(b * S_LEN + kb * 64 + r) * HDIM + hd * DK);
            unsigned int u = src[c2];
            Vt[c2 * 2 + 0][r] = ((const __bf16*)&u)[0];
            Vt[c2 * 2 + 1][r] = ((const __bf16*)&u)[1];
        }
        if (wave == 0) __builtin_amdgcn_s_wait_tensorcnt(0);
        __syncthreads();

        // ---- scores: S(16x64) = Q(16x64) x K^T, 4 n-tiles x 2 k-steps ----
        v8f sc[4];
#pragma unroll
        for (int nt = 0; nt < 4; ++nt) sc[nt] = vzero;
#pragma unroll
        for (int nt = 0; nt < 4; ++nt) {
#pragma unroll
            for (int kt = 0; kt < 2; ++kt) {
                Frag kf;
                int key = nt * 16 + l16;   // B col = key; B(k,n)=K[key][d]
#pragma unroll
                for (int vv = 0; vv < 8; ++vv) {
                    int kk = kt * 32 + (hh << 4) + (vv << 1);
                    kf.u[vv] = *(const unsigned int*)&Ks[key][kk];
                }
                sc[nt] = wmma_bf16(qa[kt], kf, sc[nt]);
            }
        }

        const float scale = 0.125f;        // 1/sqrt(64)
        bool mk[4];
#pragma unroll
        for (int nt = 0; nt < 4; ++nt)
            mk[nt] = mask[b * S_LEN + kb * 64 + nt * 16 + l16] != 0;

        // ---- online softmax per row (row = r + 8*hh within wave strip) ----
#pragma unroll
        for (int r = 0; r < 8; ++r) {
            float s[4];
#pragma unroll
            for (int nt = 0; nt < 4; ++nt)
                s[nt] = mk[nt] ? sc[nt][r] * scale : -1e9f;
            float mx = fmaxf(fmaxf(s[0], s[1]), fmaxf(s[2], s[3]));
#pragma unroll
            for (int off = 1; off < 16; off <<= 1)
                mx = fmaxf(mx, __shfl_xor(mx, off, 32));
            float nm   = fmaxf(m_i[r], mx);
            float corr = __expf(m_i[r] - nm);
            float p[4];
            float rs = 0.f;
#pragma unroll
            for (int nt = 0; nt < 4; ++nt) { p[nt] = __expf(s[nt] - nm); rs += p[nt]; }
#pragma unroll
            for (int off = 1; off < 16; off <<= 1)
                rs += __shfl_xor(rs, off, 32);
            l_i[r] = l_i[r] * corr + rs;
            m_i[r] = nm;
#pragma unroll
            for (int t = 0; t < 4; ++t) o[t][r] *= corr;
            int prow = r + 8 * hh;
#pragma unroll
            for (int nt = 0; nt < 4; ++nt)
                Ps[wave][prow][nt * 16 + l16] = (__bf16)p[nt];
        }
        asm volatile("s_wait_dscnt 0x0" ::: "memory");  // wave-local LDS RAW

        // ---- P(16x64) x V(64x64): relayout P, 2 k-steps x 4 n-tiles ----
        Frag pa[2];
#pragma unroll
        for (int kt = 0; kt < 2; ++kt)
#pragma unroll
            for (int vv = 0; vv < 8; ++vv) {
                int kk = kt * 32 + (hh << 4) + (vv << 1);
                pa[kt].u[vv] = *(const unsigned int*)&Ps[wave][l16][kk];
            }
#pragma unroll
        for (int nt = 0; nt < 4; ++nt) {
#pragma unroll
            for (int kt = 0; kt < 2; ++kt) {
                Frag vf;
                int n = nt * 16 + l16;     // B col = dk; B(k,n)=V[key][dk]
#pragma unroll
                for (int vv = 0; vv < 8; ++vv) {
                    int kk = kt * 32 + (hh << 4) + (vv << 1);
                    vf.u[vv] = *(const unsigned int*)&Vt[n][kk];
                }
                o[nt] = wmma_bf16(pa[kt], vf, o[nt]);
            }
        }
    }

    // ---- epilogue: O /= l, write concat (B,S,H*Dk) as bf16 ----
#pragma unroll
    for (int r = 0; r < 8; ++r) {
        float inv = (l_i[r] > 0.f) ? (1.0f / l_i[r]) : 0.f;
        int row = qb * 128 + wave * 16 + r + 8 * hh;
#pragma unroll
        for (int nt = 0; nt < 4; ++nt) {
            int dk = nt * 16 + l16;
            concat[(size_t)(b * S_LEN + row) * HDIM + hd * DK + dk] =
                (__bf16)(o[nt][r] * inv);
        }
    }
}

// ---------------------------------------------------------------------------
extern "C" void kernel_launch(void* const* d_in, const int* in_sizes, int n_in,
                              void* d_out, int out_size, void* d_ws, size_t ws_size,
                              hipStream_t stream)
{
    const float* q    = (const float*)d_in[0];
    const float* k    = (const float*)d_in[1];
    const float* v    = (const float*)d_in[2];
    const int*   mask = (const int*)  d_in[3];
    const float* Wq   = (const float*)d_in[4];
    const float* bq   = (const float*)d_in[5];
    const float* Wk   = (const float*)d_in[6];
    const float* bk   = (const float*)d_in[7];
    const float* Wv   = (const float*)d_in[8];
    const float* bv   = (const float*)d_in[9];
    const float* Wo   = (const float*)d_in[10];
    const float* bo   = (const float*)d_in[11];
    float* out = (float*)d_out;

    const int M = 4 * S_LEN;    // 8192 rows
    const int N = HDIM;         // 1024
    const int K = 1024;

    char* ws = (char*)d_ws;
    const size_t bytes_per = (size_t)M * N * sizeof(__bf16);  // 16 MB
    __bf16* qhb = (__bf16*)(ws + 0 * bytes_per);
    __bf16* khb = (__bf16*)(ws + 1 * bytes_per);
    __bf16* vhb = (__bf16*)(ws + 2 * bytes_per);
    __bf16* ccb = (__bf16*)(ws + 3 * bytes_per);

    dim3 gridG(N / 128, M / 128);   // (8, 64)
    gemm_bias_kernel<false, true><<<gridG, 256, 0, stream>>>(q, Wq, bq, qhb, M, N, K);
    gemm_bias_kernel<false, true><<<gridG, 256, 0, stream>>>(k, Wk, bk, khb, M, N, K);
    gemm_bias_kernel<false, true><<<gridG, 256, 0, stream>>>(v, Wv, bv, vhb, M, N, K);

    flash_attn_kernel<<<dim3(S_LEN / 128, NHEADS, 4), 256, 0, stream>>>(
        qhb, khb, vhb, mask, ccb);

    gemm_bias_kernel<true, false><<<gridG, 256, 0, stream>>>(ccb, Wo, bo, out, M, N, K);
}
